// Attention_24361054503404
// MI455X (gfx1250) — matrix operand order repaired and checked by reading, not compiled, not run on hardware
//
#include <hip/hip_runtime.h>
#include <hip/hip_bf16.h>
#include <stdint.h>

typedef __attribute__((ext_vector_type(16))) _Float16 v16h;
typedef __attribute__((ext_vector_type(8)))  float    v8f;
typedef __attribute__((ext_vector_type(4)))  unsigned int u32x4;
typedef __attribute__((ext_vector_type(8)))  int i32x8;
typedef __attribute__((ext_vector_type(4)))  int i32x4;

#define DEV __device__ __forceinline__

constexpr int BB = 2, NN = 1024, LL = 4096, CC = 256, HH = 8, DD = 32;
constexpr float SCALE = 0.17677669529663687f; // 32^-0.5

// ---- WMMA operand loaders (wave32 layouts per CDNA5 ISA 7.12.2) ----
// A-matrix f16 16x32: lane m=lane&15; element e maps K=(e<8?e:e+8)+(hi?8:0)
// => per lane: 8 contiguous halves at +0 and 8 contiguous halves at +16.
DEV v16h load16h_split(const _Float16* p) {
  union { uint4 u; _Float16 h[8]; } t0, t1;
  t0.u = *(const uint4*)(p);
  t1.u = *(const uint4*)(p + 16);
  v16h r;
#pragma unroll
  for (int e = 0; e < 8; ++e) { r[e] = t0.h[e]; r[e + 8] = t1.h[e]; }
  return r;
}

// B-matrix f16 32x16: lane n=lane&15; element e maps K=(e<8?e:e+8)+(hi?8:0)
// (caller applies the +8 offset). 8 contiguous halves at +0 and 8 at +16.
DEV v16h load16h_contig(const _Float16* p) {
  union { uint4 u; _Float16 h[8]; } t0, t1;
  t0.u = *(const uint4*)(p);
  t1.u = *(const uint4*)(p + 16);
  v16h r;
#pragma unroll
  for (int e = 0; e < 8; ++e) { r[e] = t0.h[e]; r[e + 8] = t1.h[e]; }
  return r;
}

// A-matrix from f32 memory (convert to f16): same split pattern.
DEV v16h load_a_f32(const float* p) {
  const float4 x0 = *(const float4*)(p);
  const float4 x1 = *(const float4*)(p + 4);
  const float4 x2 = *(const float4*)(p + 16);
  const float4 x3 = *(const float4*)(p + 20);
  v16h r;
  r[0] = (_Float16)x0.x; r[1] = (_Float16)x0.y; r[2] = (_Float16)x0.z; r[3] = (_Float16)x0.w;
  r[4] = (_Float16)x1.x; r[5] = (_Float16)x1.y; r[6] = (_Float16)x1.z; r[7] = (_Float16)x1.w;
  r[8] = (_Float16)x2.x; r[9] = (_Float16)x2.y; r[10] = (_Float16)x2.z; r[11] = (_Float16)x2.w;
  r[12] = (_Float16)x3.x; r[13] = (_Float16)x3.y; r[14] = (_Float16)x3.z; r[15] = (_Float16)x3.w;
  return r;
}

DEV v8f wmma_f16(v16h a, v16h b, v8f c) {
  return __builtin_amdgcn_wmma_f32_16x16x32_f16(false, a, false, b, (short)0, c,
                                                false, false);
}

// ---- Tensor Data Mover: 2D f16 tile -> LDS (D# per ISA §8.3/8.4) ----------
// This toolchain exposes the 6-arg builtin: (g0, g1, g2, g3, g_extra, cpol).
DEV void tdm_load_2d_f16(unsigned lds_off, const _Float16* gptr,
                         unsigned tile_d0, unsigned tile_d1, unsigned stride_d0) {
  const unsigned long long ga = (unsigned long long)(uintptr_t)gptr;
  u32x4 g0;
  g0[0] = 1u;                                        // count=1, user descriptor
  g0[1] = lds_off;                                   // lds_addr (bytes)
  g0[2] = (unsigned)ga;                              // global_addr[31:0]
  g0[3] = (unsigned)((ga >> 32) & 0x01FFFFFFu) | (2u << 30);  // addr hi | type=2
  i32x8 g1;
  g1[0] = (int)(1u << 16);                           // data_size=1 (2B), no mcast
  g1[1] = (int)((tile_d0 & 0xFFFFu) << 16);          // tensor_dim0 = tile_d0
  g1[2] = (int)((tile_d1 & 0xFFFFu) << 16);          // tensor_dim1 = tile_d1
  g1[3] = (int)((tile_d0 & 0xFFFFu) << 16);          // tile_dim0
  g1[4] = (int)(tile_d1 & 0xFFFFu);                  // tile_dim1 (tile_dim2=0)
  g1[5] = (int)stride_d0;                            // tensor_dim0_stride lo
  g1[6] = 0;
  g1[7] = 0;
  const i32x4 z4 = {0, 0, 0, 0};                     // groups 2/3 unused (2D)
  const i32x8 z8 = {0, 0, 0, 0, 0, 0, 0, 0};
  __builtin_amdgcn_tensor_load_to_lds(g0, g1, z4, z4, z8, 0);
}

// ---- weight pre-pass: W f32 [C,C] row-major -> Wt f16 col-major -----------
__global__ __launch_bounds__(256)
void transpose_w_kernel(const float* __restrict__ W, _Float16* __restrict__ Wt) {
  const int idx = blockIdx.x * 256 + threadIdx.x;   // over C*C
  const int k = idx / CC, n = idx % CC;
  Wt[(size_t)n * CC + k] = (_Float16)W[idx];
}

// ---------------- input projections: P = X @ W  (f32 in -> f16 out) --------
// mode 0: store [rows, C] row-major (Q, K)
// mode 1: store V transposed per head: Out[((b*H+h)*D + d)*L + l]
__global__ __launch_bounds__(32)
void proj_qkv_kernel(const float* __restrict__ X, const _Float16* __restrict__ Wt,
                     _Float16* __restrict__ Out, int mode) {
  const int lane = threadIdx.x;
  const int row0 = blockIdx.x * 16;
  const int col0 = blockIdx.y * 16;
  const int mlan = lane & 15;
  const int hi   = lane >> 4;
  const int k0   = hi ? 8 : 0;
  v8f acc = {};
#pragma unroll
  for (int kk = 0; kk < CC; kk += 32) {
    v16h a = load_a_f32(X + (size_t)(row0 + mlan) * CC + kk + k0);
    v16h b = load16h_contig(Wt + (size_t)(col0 + mlan) * CC + kk + (hi ? 8 : 0));
    acc = wmma_f16(a, b, acc);
  }
  if (mode == 0) {
#pragma unroll
    for (int i = 0; i < 8; ++i) {
      const int m = i + hi * 8;
      Out[(size_t)(row0 + m) * CC + col0 + mlan] = (_Float16)acc[i];
    }
  } else {
    const int h = col0 >> 5;
    const int d = (col0 & 31) + mlan;
    const int b = row0 / LL;
    const int lb = row0 - b * LL;
#pragma unroll
    for (int i = 0; i < 8; ++i) {
      const int l = lb + i + hi * 8;
      Out[(((size_t)b * HH + h) * DD + d) * LL + l] = (_Float16)acc[i];
    }
  }
}

// ---------- fused flash attention + per-(n,l) cross-head MLP mask ----------
__global__ __launch_bounds__(32)
void fused_attn_kernel(const _Float16* __restrict__ Qh, const _Float16* __restrict__ Kh,
                       const _Float16* __restrict__ Vt,
                       const float* __restrict__ Wl1, const float* __restrict__ bl1,
                       const float* __restrict__ Wl2, const float* __restrict__ bl2,
                       _Float16* __restrict__ Xh, float* __restrict__ maskout) {
  const int lane = threadIdx.x;
  const int nt = NN / 16;
  const int b = blockIdx.x / nt;
  const int n0 = (blockIdx.x % nt) * 16;
  const int mlan = lane & 15;
  const int hi = lane >> 4;
  const int k0 = hi ? 8 : 0;

  // double-buffered TDM staging tiles + P re-layout bounce
  __shared__ __align__(128) _Float16 lds_k[2][32 * CC];      // K rows  [l][c]
  __shared__ __align__(128) _Float16 lds_v[2][HH * DD * 32]; // V^T     [h*D+d][l]
  __shared__ __align__(128) _Float16 lds_p[16 * 32];

  // MLP params (uniform addresses -> scalar loads / SGPR operands)
  float wl1[HH * HH], vbl1[HH], wl2[HH];
#pragma unroll
  for (int i = 0; i < HH * HH; ++i) wl1[i] = Wl1[i];
#pragma unroll
  for (int i = 0; i < HH; ++i) { vbl1[i] = bl1[i]; wl2[i] = Wl2[i]; }
  const float vbl2 = bl2[0];

  // resident Q tiles, one A-matrix per head (K-dim = D = 32)
  v16h qa[HH];
#pragma unroll
  for (int h = 0; h < HH; ++h)
    qa[h] = load16h_split(Qh + ((size_t)b * NN + n0 + mlan) * CC + h * DD + k0);

  v8f o0[HH], o1[HH];              // O accumulators, d=0..15 / 16..31
  float mrow[HH][8], rrow[HH][8];  // online softmax stats (C-layout slots)
  v8f vzero = {};
#pragma unroll
  for (int h = 0; h < HH; ++h) {
    o0[h] = vzero; o1[h] = vzero;
#pragma unroll
    for (int i = 0; i < 8; ++i) { mrow[h][i] = -1e30f; rrow[h][i] = 0.f; }
  }

  auto issue_tiles = [&](int buf, int l0s) {
    // K tile: 32 rows x 256 cols, row stride C
    tdm_load_2d_f16((unsigned)(uintptr_t)&lds_k[buf][0],
                    Kh + ((size_t)b * LL + l0s) * CC, CC, 32, CC);
    // V^T tile: 256 rows (h*D+d) x 32 cols (l), row stride L
    tdm_load_2d_f16((unsigned)(uintptr_t)&lds_v[buf][0],
                    Vt + ((size_t)b * HH * DD) * LL + l0s, 32, HH * DD, LL);
  };

  issue_tiles(0, 0);  // prologue
  int cur = 0;

  for (int l0 = 0; l0 < LL; l0 += 32) {
    const int nxt = cur ^ 1;
    if (l0 + 32 < LL) {
      issue_tiles(nxt, l0 + 32);
      __builtin_amdgcn_s_wait_tensorcnt(2);  // in-order: current tile resident
    } else {
      __builtin_amdgcn_s_wait_tensorcnt(0);
    }
    const _Float16* kb_base = &lds_k[cur][0];
    const _Float16* vb_base = &lds_v[cur][0];

    // scores for all 8 heads: S = q . k^T * scale (one WMMA per 16x16 tile)
    v8f s0[HH], s1[HH];
#pragma unroll
    for (int h = 0; h < HH; ++h) {
      const _Float16* kb = kb_base + (size_t)mlan * CC + h * DD + (hi ? 8 : 0);
      v16h bk0 = load16h_contig(kb);
      v16h bk1 = load16h_contig(kb + (size_t)16 * CC);
      s0[h] = wmma_f16(qa[h], bk0, vzero);
      s1[h] = wmma_f16(qa[h], bk1, vzero);
#pragma unroll
      for (int i = 0; i < 8; ++i) { s0[h][i] *= SCALE; s1[h][i] *= SCALE; }
    }

    // cross-head MLP mask on the scaled scores; element (n0+m, l0+nl(+16))
#pragma unroll
    for (int i = 0; i < 8; ++i) {
      const int m = i + hi * 8;
      float fe0[HH], fe1[HH];
#pragma unroll
      for (int j = 0; j < HH; ++j) {
        float a0 = vbl1[j], a1 = vbl1[j];
#pragma unroll
        for (int h = 0; h < HH; ++h) {
          a0 = fmaf(s0[h][i], wl1[h * HH + j], a0);
          a1 = fmaf(s1[h][i], wl1[h * HH + j], a1);
        }
        fe0[j] = fmaxf(a0, 0.f);
        fe1[j] = fmaxf(a1, 0.f);
      }
      float m0 = vbl2, m1 = vbl2;
#pragma unroll
      for (int j = 0; j < HH; ++j) {
        m0 = fmaf(fe0[j], wl2[j], m0);
        m1 = fmaf(fe1[j], wl2[j], m1);
      }
      const size_t base = ((size_t)b * NN + n0 + m) * LL + l0 + mlan;
      maskout[base] = fmaxf(m0, 0.f);
      maskout[base + 16] = fmaxf(m1, 0.f);
    }

    // online softmax + P@V per head
#pragma unroll
    for (int h = 0; h < HH; ++h) {
      float corr[8];
#pragma unroll
      for (int i = 0; i < 8; ++i) {
        float t = fmaxf(s0[h][i], s1[h][i]);
#pragma unroll
        for (int off = 8; off >= 1; off >>= 1)  // 16-lane half reduction
          t = fmaxf(t, __shfl_xor(t, off, 32));
        const float mnew = fmaxf(mrow[h][i], t);
        const float c = __expf(mrow[h][i] - mnew);
        const float p0v = __expf(s0[h][i] - mnew);
        const float p1v = __expf(s1[h][i] - mnew);
        s0[h][i] = p0v; s1[h][i] = p1v;
        float rs = p0v + p1v;
#pragma unroll
        for (int off = 8; off >= 1; off >>= 1) rs += __shfl_xor(rs, off, 32);
        rrow[h][i] = rrow[h][i] * c + rs;
        mrow[h][i] = mnew;
        corr[i] = c;
      }
#pragma unroll
      for (int i = 0; i < 8; ++i) { o0[h][i] *= corr[i]; o1[h][i] *= corr[i]; }

      // bounce P (C-layout) through LDS, reload in A-layout (16x32, K = l)
#pragma unroll
      for (int i = 0; i < 8; ++i) {
        const int m = i + hi * 8;
        lds_p[m * 32 + mlan] = (_Float16)s0[h][i];
        lds_p[m * 32 + mlan + 16] = (_Float16)s1[h][i];
      }
      v16h pa = load16h_split(&lds_p[mlan * 32 + k0]);

      const _Float16* vb = vb_base + (size_t)(h * DD + mlan) * 32 + (hi ? 8 : 0);
      v16h bv0 = load16h_contig(vb);
      v16h bv1 = load16h_contig(vb + (size_t)16 * 32);
      o0[h] = wmma_f16(pa, bv0, o0[h]);
      o1[h] = wmma_f16(pa, bv1, o1[h]);
    }
    cur = nxt;
  }

  // normalize and emit per-head context rows as f16 [B,N,C]
#pragma unroll
  for (int h = 0; h < HH; ++h) {
#pragma unroll
    for (int i = 0; i < 8; ++i) {
      const float inv = 1.f / rrow[h][i];
      const int m = i + hi * 8;
      const size_t base = ((size_t)b * NN + n0 + m) * CC + h * DD;
      Xh[base + mlan] = (_Float16)(o0[h][i] * inv);
      Xh[base + 16 + mlan] = (_Float16)(o1[h][i] * inv);
    }
  }
}

// ---------------- output projection: out = Xh @ Wp + bp (f32 out) ----------
__global__ __launch_bounds__(32)
void proj_out_kernel(const _Float16* __restrict__ Xh, const _Float16* __restrict__ Wpt,
                     const float* __restrict__ bp, float* __restrict__ Out) {
  const int lane = threadIdx.x;
  const int row0 = blockIdx.x * 16;
  const int col0 = blockIdx.y * 16;
  const int mlan = lane & 15;
  const int hi = lane >> 4;
  const int k0 = hi ? 8 : 0;
  v8f acc = {};
#pragma unroll
  for (int kk = 0; kk < CC; kk += 32) {
    v16h a = load16h_split(Xh + (size_t)(row0 + mlan) * CC + kk + k0);
    v16h b = load16h_contig(Wpt + (size_t)(col0 + mlan) * CC + kk + (hi ? 8 : 0));
    acc = wmma_f16(a, b, acc);
  }
  const float bias = bp[col0 + mlan];
#pragma unroll
  for (int i = 0; i < 8; ++i) {
    const int m = i + hi * 8;
    Out[(size_t)(row0 + m) * CC + col0 + mlan] = acc[i] + bias;
  }
}

extern "C" void kernel_launch(void* const* d_in, const int* in_sizes, int n_in,
                              void* d_out, int out_size, void* d_ws, size_t ws_size,
                              hipStream_t stream) {
  const float* query = (const float*)d_in[0];
  const float* key   = (const float*)d_in[1];
  const float* value = (const float*)d_in[2];
  // d_in[3] key_padding_mask (unused), d_in[4] hw_lvl (unused)
  const float* Wq = (const float*)d_in[5];
  const float* Wk = (const float*)d_in[6];
  const float* Wv = (const float*)d_in[7];
  const float* Wp = (const float*)d_in[8];
  const float* bp = (const float*)d_in[9];
  const float* Wl1 = (const float*)d_in[10];
  const float* bl1 = (const float*)d_in[11];
  const float* Wl2 = (const float*)d_in[12];
  const float* bl2 = (const float*)d_in[13];

  float* out = (float*)d_out;                   // x: B*N*C floats
  float* maskout = out + (size_t)BB * NN * CC;  // mask: B*N*L floats

  _Float16* Qh = (_Float16*)d_ws;               // [B,N,C]   f16, 1 MB
  _Float16* Kh = Qh + (size_t)BB * NN * CC;     // [B,L,C]   f16, 4 MB
  _Float16* Vt = Kh + (size_t)BB * LL * CC;     // [B,H,D,L] f16, 4 MB
  _Float16* Xh = Vt + (size_t)BB * LL * CC;     // [B,N,C]   f16, 1 MB
  _Float16* WtQ = Xh + (size_t)BB * NN * CC;    // 4x [C,C] f16 col-major
  _Float16* WtK = WtQ + (size_t)CC * CC;
  _Float16* WtV = WtK + (size_t)CC * CC;
  _Float16* WtP = WtV + (size_t)CC * CC;

  const int wgrid = CC * CC / 256;
  transpose_w_kernel<<<wgrid, 256, 0, stream>>>(Wq, WtQ);
  transpose_w_kernel<<<wgrid, 256, 0, stream>>>(Wk, WtK);
  transpose_w_kernel<<<wgrid, 256, 0, stream>>>(Wv, WtV);
  transpose_w_kernel<<<wgrid, 256, 0, stream>>>(Wp, WtP);

  proj_qkv_kernel<<<dim3(BB * NN / 16, CC / 16), 32, 0, stream>>>(query, WtQ, Qh, 0);
  proj_qkv_kernel<<<dim3(BB * LL / 16, CC / 16), 32, 0, stream>>>(key, WtK, Kh, 0);
  proj_qkv_kernel<<<dim3(BB * LL / 16, CC / 16), 32, 0, stream>>>(value, WtV, Vt, 1);
  fused_attn_kernel<<<dim3(BB * NN / 16), 32, 0, stream>>>(Qh, Kh, Vt, Wl1, bl1,
                                                           Wl2, bl2, Xh, maskout);
  proj_out_kernel<<<dim3(BB * NN / 16, CC / 16), 32, 0, stream>>>(Xh, WtP, bp, out);
}
